// TemporalAttention_11476152615095
// MI455X (gfx1250) — matrix operand order repaired and checked by reading, not compiled, not run on hardware
//
#include <hip/hip_runtime.h>
#include <hip/hip_bf16.h>

typedef __attribute__((ext_vector_type(16))) __bf16 v16bf;
typedef __attribute__((ext_vector_type(8)))  float  v8f;

// Problem constants
constexpr int kT  = 64;    // sequence length
constexpr int kD  = 256;   // embed dim
constexpr int kHD = 32;    // head dim
// LDS strides (bf16 elements), padded to avoid bank conflicts
constexpr int XS = 264;    // 64 x 256 tiles (x/ctx, Q, K)
constexpr int VS = 72;     // V^T  [256][64+pad]
constexpr int PS = 72;     // per-wave P scratch [16][64+pad]

union FragB16 { unsigned int u[8]; v16bf v; };
union FragF   { float f[8];        v8f   v; };

__device__ __forceinline__ unsigned short f2bf(float x) {
  // round-to-nearest-even f32 -> bf16 (bits)
  unsigned u = __float_as_uint(x);
  unsigned r = u + 0x7FFFu + ((u >> 16) & 1u);
  return (unsigned short)(r >> 16);
}

__global__ void cvt_bf16(const float* __restrict__ src,
                         unsigned short* __restrict__ dst, int n) {
  int i = blockIdx.x * blockDim.x + threadIdx.x;
  if (i < n) dst[i] = f2bf(src[i]);
}

__global__ void __launch_bounds__(256)
temporal_attn_kernel(const float* __restrict__ x,
                     const unsigned short* __restrict__ wq,  // bf16 bits, [o][d]
                     const unsigned short* __restrict__ wk,
                     const unsigned short* __restrict__ wv,
                     const unsigned short* __restrict__ wo,
                     const float* __restrict__ bq, const float* __restrict__ bk,
                     const float* __restrict__ bv, const float* __restrict__ bo,
                     const float* __restrict__ tb,            // [8][64][64]
                     float* __restrict__ out) {
  extern __shared__ unsigned short smem[];
  unsigned short* xb  = smem;               // x tile, later reused as ctx
  unsigned short* qb  = smem + 1 * kT * XS;
  unsigned short* kb  = smem + 2 * kT * XS;
  unsigned short* vbT = smem + 3 * kT * XS; // [256][VS] (V transposed)
  unsigned short* pb  = vbT + kD * VS;      // [8 waves][16][PS]

  const int tid  = threadIdx.x;
  const int wave = tid >> 5;
  const int lane = tid & 31;
  const int l15  = lane & 15;
  const int hi   = lane >> 4;               // half-wave select
  const long seq = blockIdx.x;
  const float* xs = x + seq * (long)(kT * kD);

  // ---- Phase 0: load x tile -> LDS (bf16) -------------------------------
  for (int r = 0; r < kT; ++r)
    xb[r * XS + tid] = f2bf(xs[r * kD + tid]);
  __syncthreads();

  // ---- Phase 1: Q/K/V projections (y = x @ W^T + b) ---------------------
  const float qscale = 0.17677669529663687f; // 1/sqrt(32)
  for (int p = 0; p < 3; ++p) {
    const unsigned short* W = (p == 0) ? wq : (p == 1) ? wk : wv;
    const float* bias       = (p == 0) ? bq : (p == 1) ? bk : bv;
    for (int t = wave; t < 64; t += 8) {
      const int mb = (t >> 4) * 16;   // token block
      const int nb = (t & 15) * 16;   // output-dim block
      FragF acc;
      #pragma unroll
      for (int j = 0; j < 8; ++j) acc.f[j] = 0.f;
      #pragma unroll
      for (int kk = 0; kk < kD; kk += 32) {
        FragB16 a, b;
        #pragma unroll
        for (int j = 0; j < 8; ++j) {
          const int ka  = kk + ((j < 4) ? 0 : 16) + hi * 8 + 2 * (j & 3);
          a.u[j] = *(const unsigned int*)&xb[(mb + l15) * XS + ka];
          const int kb2 = kk + ((j < 4) ? 0 : 16) + hi * 8 + 2 * (j & 3);
          b.u[j] = *(const unsigned int*)&W[(nb + l15) * kD + kb2];
        }
        acc.v = __builtin_amdgcn_wmma_f32_16x16x32_bf16(
            false, a.v, false, b.v, (short)0, acc.v, false, false);
      }
      const float bv_ = bias[nb + l15];
      #pragma unroll
      for (int j = 0; j < 8; ++j) acc.f[j] += bv_;
      if (p == 0) {
        #pragma unroll
        for (int j = 0; j < 8; ++j) acc.f[j] *= qscale;
      }
      if (p == 2) { // store V transposed: vbT[dim][token]
        #pragma unroll
        for (int j = 0; j < 8; ++j)
          vbT[(nb + l15) * VS + mb + hi * 8 + j] = f2bf(acc.f[j]);
      } else {
        unsigned short* dst = (p == 0) ? qb : kb;
        #pragma unroll
        for (int j = 0; j < 8; ++j)
          dst[(mb + hi * 8 + j) * XS + nb + l15] = f2bf(acc.f[j]);
      }
    }
  }
  __syncthreads();

  // ---- Phase 2: attention, one head per wave ----------------------------
  {
    const int h = wave;
    const float* tbh = tb + h * (kT * kT);
    unsigned short* pw = pb + h * 16 * PS;
    for (int mb = 0; mb < kT; mb += 16) {
      // A fragment: Q rows [mb, mb+16), dims [h*32, h*32+32)
      FragB16 aq;
      #pragma unroll
      for (int j = 0; j < 8; ++j) {
        const int ka = h * kHD + ((j < 4) ? 0 : 16) + hi * 8 + 2 * (j & 3);
        aq.u[j] = *(const unsigned int*)&qb[(mb + l15) * XS + ka];
      }
      // scores S = Q_h K_h^T * scale + bias  (16 x 64)
      FragF s[4];
      #pragma unroll
      for (int nt = 0; nt < 4; ++nt) {
        FragB16 b;
        #pragma unroll
        for (int j = 0; j < 8; ++j) {
          const int kk2 = h * kHD + ((j < 4) ? 0 : 16) + hi * 8 + 2 * (j & 3);
          b.u[j] = *(const unsigned int*)&kb[(nt * 16 + l15) * XS + kk2];
        }
        v8f z = {};
        s[nt].v = __builtin_amdgcn_wmma_f32_16x16x32_bf16(
            false, aq.v, false, b.v, (short)0, z, false, false);
        #pragma unroll
        for (int j = 0; j < 8; ++j)
          s[nt].f[j] += tbh[(mb + hi * 8 + j) * kT + nt * 16 + l15];
      }
      // row softmax: rows live across 16-lane halves (C layout)
      #pragma unroll
      for (int j = 0; j < 8; ++j) {
        float mx = fmaxf(fmaxf(s[0].f[j], s[1].f[j]),
                         fmaxf(s[2].f[j], s[3].f[j]));
        #pragma unroll
        for (int off = 8; off >= 1; off >>= 1)
          mx = fmaxf(mx, __shfl_xor(mx, off, 32));
        float sum = 0.f;
        #pragma unroll
        for (int nt = 0; nt < 4; ++nt) {
          s[nt].f[j] = __expf(s[nt].f[j] - mx);
          sum += s[nt].f[j];
        }
        #pragma unroll
        for (int off = 8; off >= 1; off >>= 1)
          sum += __shfl_xor(sum, off, 32);
        const float inv = 1.0f / sum;
        #pragma unroll
        for (int nt = 0; nt < 4; ++nt) s[nt].f[j] *= inv;
      }
      // P (16x64) -> per-wave LDS scratch, bf16
      #pragma unroll
      for (int nt = 0; nt < 4; ++nt)
        #pragma unroll
        for (int j = 0; j < 8; ++j)
          pw[(hi * 8 + j) * PS + nt * 16 + l15] = f2bf(s[nt].f[j]);
      asm volatile("s_wait_dscnt 0" ::: "memory"); // wave-local LDS visibility

      // ctx = P @ V_h : (16x64)(64x32) -> 16x32 (2 tiles of 16)
      #pragma unroll
      for (int nt2 = 0; nt2 < 2; ++nt2) {
        FragF c;
        #pragma unroll
        for (int j = 0; j < 8; ++j) c.f[j] = 0.f;
        #pragma unroll
        for (int ks = 0; ks < 2; ++ks) {
          FragB16 a, b;
          #pragma unroll
          for (int j = 0; j < 8; ++j) {
            const int ka  = ks * 32 + ((j < 4) ? 0 : 16) + hi * 8 + 2 * (j & 3);
            a.u[j] = *(const unsigned int*)&pw[l15 * PS + ka];
            const int kk2 = ks * 32 + ((j < 4) ? 0 : 16) + hi * 8 + 2 * (j & 3);
            b.u[j] = *(const unsigned int*)
                &vbT[(h * kHD + nt2 * 16 + l15) * VS + kk2];
          }
          c.v = __builtin_amdgcn_wmma_f32_16x16x32_bf16(
              false, a.v, false, b.v, (short)0, c.v, false, false);
        }
        // ctx -> xb (reused): [token][h*32 + nt2*16 + col]
        #pragma unroll
        for (int j = 0; j < 8; ++j)
          xb[(mb + hi * 8 + j) * XS + h * kHD + nt2 * 16 + l15] =
              f2bf(c.f[j]);
      }
    }
  }
  __syncthreads();

  // ---- Phase 3: output projection out = ctx @ Wo^T + bo -----------------
  float* op = out + seq * (long)(kT * kD);
  for (int t = wave; t < 64; t += 8) {
    const int mb = (t >> 4) * 16;
    const int nb = (t & 15) * 16;
    FragF acc;
    #pragma unroll
    for (int j = 0; j < 8; ++j) acc.f[j] = 0.f;
    #pragma unroll
    for (int kk = 0; kk < kD; kk += 32) {
      FragB16 a, b;
      #pragma unroll
      for (int j = 0; j < 8; ++j) {
        const int ka  = kk + ((j < 4) ? 0 : 16) + hi * 8 + 2 * (j & 3);
        a.u[j] = *(const unsigned int*)&xb[(mb + l15) * XS + ka];
        const int kb2 = kk + ((j < 4) ? 0 : 16) + hi * 8 + 2 * (j & 3);
        b.u[j] = *(const unsigned int*)&wo[(nb + l15) * kD + kb2];
      }
      acc.v = __builtin_amdgcn_wmma_f32_16x16x32_bf16(
          false, a.v, false, b.v, (short)0, acc.v, false, false);
    }
    const float bo_ = bo[nb + l15];
    #pragma unroll
    for (int j = 0; j < 8; ++j)
      op[(mb + hi * 8 + j) * kD + nb + l15] = acc.f[j] + bo_;
  }
}

extern "C" void kernel_launch(void* const* d_in, const int* in_sizes, int n_in,
                              void* d_out, int out_size, void* d_ws,
                              size_t ws_size, hipStream_t stream) {
  const float* x  = (const float*)d_in[0];
  const float* Wq = (const float*)d_in[1];
  const float* bq = (const float*)d_in[2];
  const float* Wk = (const float*)d_in[3];
  const float* bk = (const float*)d_in[4];
  const float* Wv = (const float*)d_in[5];
  const float* bv = (const float*)d_in[6];
  const float* Wo = (const float*)d_in[7];
  const float* bo = (const float*)d_in[8];
  const float* tb = (const float*)d_in[9];
  float* out = (float*)d_out;

  // Workspace: 4 bf16 weight matrices, 128KB each
  unsigned short* ws = (unsigned short*)d_ws;
  unsigned short* wq = ws;
  unsigned short* wk = ws + 1 * 65536;
  unsigned short* wv = ws + 2 * 65536;
  unsigned short* wo = ws + 3 * 65536;
  const int WN = kD * kD;
  cvt_bf16<<<WN / 256, 256, 0, stream>>>(Wq, wq, WN);
  cvt_bf16<<<WN / 256, 256, 0, stream>>>(Wk, wk, WN);
  cvt_bf16<<<WN / 256, 256, 0, stream>>>(Wv, wv, WN);
  cvt_bf16<<<WN / 256, 256, 0, stream>>>(Wo, wo, WN);

  const int nseq = in_sizes[0] / (kT * kD); // 8*170 = 1360 sequences
  const size_t lds_bytes =
      (size_t)(3 * kT * XS + kD * VS + 8 * 16 * PS) * sizeof(unsigned short);
  temporal_attn_kernel<<<nseq, 256, lds_bytes, stream>>>(
      x, wq, wk, wv, wo, bq, bk, bv, bo, tb, out);
}
